// DeformLayer_8778913153618
// MI455X (gfx1250) — matrix operand order repaired and checked by reading, not compiled, not run on hardware
//
#include <hip/hip_runtime.h>
#include <hip/hip_bf16.h>
#include <math.h>
#include <stdint.h>

// ---------------- CDNA5 WMMA plumbing (wave32, 16x16x32 bf16) ----------------
typedef __attribute__((ext_vector_type(16))) __bf16 v16bf;
typedef __attribute__((ext_vector_type(8)))  float  v8f;

union FragBf { v16bf v; unsigned u[8]; };

__device__ __forceinline__ int imin(int a, int b) { return a < b ? a : b; }
__device__ __forceinline__ int imax(int a, int b) { return a > b ? a : b; }

__device__ __forceinline__ unsigned short f2bf(float f) {
    unsigned u = __builtin_bit_cast(unsigned int, f);
    unsigned r = (u + 0x7FFFu + ((u >> 16) & 1u)) >> 16;   // round-to-nearest-even
    return (unsigned short)r;
}

// A-matrix 16x32 bf16 fragment from LDS, rows [16][lda] (lda even).
__device__ __forceinline__ void load_frag_a(const unsigned short* __restrict__ A, int lda, FragBf& fa) {
    int lane  = threadIdx.x & 31;
    int row   = lane & 15;
    int khalf = (lane >> 4) << 3;                 // +8 for lanes 16..31
    const unsigned short* p = A + row * lda;
#pragma unroll
    for (int v = 0; v < 8; ++v) {
        int kbase = ((v < 4) ? (2 * v) : (16 + 2 * (v - 4))) + khalf;
        fa.u[v] = *reinterpret_cast<const unsigned*>(p + kbase);  // packed K pair
    }
}

// B-matrix 32x16 bf16 fragment from LDS staged transposed: Bt[n][lda] (k contiguous).
__device__ __forceinline__ void load_frag_b(const unsigned short* __restrict__ Bt, int lda, FragBf& fb) {
    int lane  = threadIdx.x & 31;
    int col   = lane & 15;
    int khalf = (lane >> 4) << 3;                 // +8 for lanes 16..31
    const unsigned short* p = Bt + col * lda + khalf;
#pragma unroll
    for (int v = 0; v < 8; ++v)
        fb.u[v] = *reinterpret_cast<const unsigned*>(p + ((v < 4) ? (2 * v) : (16 + 2 * (v - 4))));
}

__device__ __forceinline__ v8f wmma_bf16(const FragBf& a, const FragBf& b, v8f c) {
    return __builtin_amdgcn_wmma_f32_16x16x32_bf16(false, a.v, false, b.v,
                                                   (short)0, c, false, false);
}

// CDNA5 async global->LDS copy of 16 bytes per lane (ASYNCcnt-tracked, no VGPR data path)
__device__ __forceinline__ void async_g2l_b128(unsigned lds_byte_off, unsigned long long gaddr) {
    asm volatile("global_load_async_to_lds_b128 %0, %1, off"
                 :: "v"(lds_byte_off), "v"(gaddr) : "memory");
}
__device__ __forceinline__ void wait_async0() {
    asm volatile("s_wait_asynccnt 0x0" ::: "memory");
}

// ---------------- Problem constants ----------------
#define BN_  8
#define CI_  256
#define CO_  256
#define HH_  64
#define WW_  64
#define HW_  4096            // 64*64
#define KDC_ 2304            // CI*9   (deform / offset conv reduction)
#define KUP_ 1024            // CI*4   (deconv per-phase reduction)
#define MTOT_ 32768          // BN*H*W
#define LDA_ 36              // LDS A row stride (ushorts)
#define LDB_ 40              // B tile row stride (ushorts): 80B rows -> 16B aligned,
                             // bank stride 20 -> conflict-free over 16 cols

// ---------------- Weight packing (fp32 -> bf16, pre-tiled B-matrix layouts) ----------------
// B1[k][32]   : offset conv, k = c*9 + p, col n = out channel (27 valid, pad 0)
__global__ void k_pack_woff(const float* __restrict__ w_off, unsigned short* __restrict__ B1) {
    int i = blockIdx.x * blockDim.x + threadIdx.x;
    if (i >= KDC_ * 32) return;
    int k = i >> 5, n = i & 31;
    B1[i] = (n < 27) ? f2bf(w_off[n * KDC_ + k]) : (unsigned short)0;
}
// B2t[kc][n][LDB_] : deform conv, k = kc*32 + kk (kk<32), pre-tiled for async->LDS
__global__ void k_pack_wdcn(const float* __restrict__ w_dcn, unsigned short* __restrict__ B2t) {
    int i = blockIdx.x * blockDim.x + threadIdx.x;
    if (i >= (KDC_ / 32) * CO_ * LDB_) return;
    int kk = i % LDB_;
    int n  = (i / LDB_) % CO_;
    int kc = i / (LDB_ * CO_);
    unsigned short v = 0;
    if (kk < 32) v = f2bf(w_dcn[n * KDC_ + kc * 32 + kk]);
    B2t[i] = v;
}
// B3t[ph][kc][o][LDB_] : transposed conv; k = c*4 + ty*2 + tx ; tap (i,j)=(2ty+pY, 2tx+pX)
// weight = flip(w_up)[c][o][i][j] = w_up[c][o][3-i][3-j]
__global__ void k_pack_wup(const float* __restrict__ w_up, unsigned short* __restrict__ B3t) {
    int i = blockIdx.x * blockDim.x + threadIdx.x;
    if (i >= 4 * (KUP_ / 32) * CO_ * LDB_) return;
    int kk = i % LDB_;
    int o  = (i / LDB_) % CO_;
    int kc = (i / (LDB_ * CO_)) % (KUP_ / 32);
    int ph = i / (LDB_ * CO_ * (KUP_ / 32));
    unsigned short v = 0;
    if (kk < 32) {
        int k = kc * 32 + kk;
        int pY = ph >> 1, pX = ph & 1;
        int c = k >> 2, t = k & 3, ty = t >> 1, tx = t & 1;
        int ii = 2 * ty + pY, jj = 2 * tx + pX;
        v = f2bf(w_up[((c * CO_ + o) * 4 + (3 - ii)) * 4 + (3 - jj)]);
    }
    B3t[i] = v;
}

// ---------------- Kernel 1: 3x3 conv 256->27 (+bias, sigmoid on mask) ----------------
// GEMM M=32768, N=32(27), K=2304.  Block: 128 rows x 32 cols, 8 waves (2 WMMA each/chunk).
__global__ __launch_bounds__(256)
void k_offset_conv(const float* __restrict__ x, const unsigned short* __restrict__ B1,
                   const float* __restrict__ b_off, float* __restrict__ om) {
    __shared__ __align__(16) unsigned short Alds[128 * LDA_];
    __shared__ __align__(16) unsigned short Btlds[32 * LDA_];
    const int tid  = threadIdx.x;
    const int wave = tid >> 5;
    const int m0   = blockIdx.x * 128;

    v8f acc0 = {}; v8f acc1 = {};
    for (int k0 = 0; k0 < KDC_; k0 += 32) {
        for (int e = tid; e < 128 * 32; e += 256) {        // A: im2col, zero-pad
            int r = e >> 5, kk = e & 31;
            int m = m0 + r;
            int b = m >> 12, hw = m & 4095;
            int h = hw >> 6, w = hw & 63;
            int k = k0 + kk;
            int c = k / 9, p = k - 9 * c;
            int h2 = h + (p / 3) - 1;
            int w2 = w + (p % 3) - 1;
            float v = 0.f;
            if (h2 >= 0 && h2 < HH_ && w2 >= 0 && w2 < WW_)
                v = x[((b * CI_ + c) * HW_) + h2 * WW_ + w2];
            Alds[r * LDA_ + kk] = f2bf(v);
        }
        for (int e = tid; e < 32 * 32; e += 256) {         // Bt (k contiguous)
            int n = e >> 5, kk = e & 31;
            const unsigned short* src = &B1[(k0 + kk) * 32 + n];
            __builtin_prefetch(src + 32 * 32, 0, 0);
            Btlds[n * LDA_ + kk] = *src;
        }
        __syncthreads();
        FragBf fa, fb0, fb1;
        load_frag_a(Alds + (wave * 16) * LDA_, LDA_, fa);
        load_frag_b(Btlds, LDA_, fb0);
        load_frag_b(Btlds + 16 * LDA_, LDA_, fb1);
        acc0 = wmma_bf16(fa, fb0, acc0);
        acc1 = wmma_bf16(fa, fb1, acc1);
        __syncthreads();
    }
    const int lane = tid & 31;
    const int n0   = lane & 15;
#pragma unroll
    for (int i = 0; i < 8; ++i) {
        int m  = m0 + wave * 16 + ((lane >> 4) * 8 + i);
        int b  = m >> 12, hw = m & 4095;
#pragma unroll
        for (int half = 0; half < 2; ++half) {
            int n = half * 16 + n0;
            if (n >= 27) continue;
            float v = (half ? acc1[i] : acc0[i]) + b_off[n];
            if (n >= 18) v = 1.f / (1.f + expf(-v));      // sigmoid(mask)
            om[(b * 27 + n) * HW_ + hw] = v;
        }
    }
}

// ---------------- Kernel 2: deformable conv + BN1 + ReLU ----------------
// GEMM M=32768, N=256 (full in one block), K=2304.  Block: 16 rows x 256 cols; grid 2048.
// A gathered once per block; B staged with global_load_async_to_lds_b128.
__global__ __launch_bounds__(256)
void k_deform(const float* __restrict__ x, const float* __restrict__ om,
              const unsigned short* __restrict__ B2t,
              const float* __restrict__ g1, const float* __restrict__ be1,
              const float* __restrict__ mu1, const float* __restrict__ va1,
              float* __restrict__ y_out, unsigned short* __restrict__ y_bf) {
    __shared__ __align__(16) unsigned short Alds[16 * LDA_];
    __shared__ __align__(16) unsigned short Btlds[256 * LDB_];
    __shared__ int   midx[16 * 9 * 4];
    __shared__ float mwt[16 * 9 * 4];
    __shared__ int   mb[16];

    const int tid  = threadIdx.x;
    const int wave = tid >> 5;
    const int lane = tid & 31;
    const int m0   = blockIdx.x * 16;

    // bilinear corner metadata for 16 spatial rows x 9 taps (mask folded in)
    for (int e = tid; e < 16 * 9; e += 256) {
        int r = e / 9, p = e - 9 * r;
        int m = m0 + r;
        int b = m >> 12, hw = m & 4095;
        int h = hw >> 6, w = hw & 63;
        if (p == 0) mb[r] = b;
        float dy = om[(b * 27 + p) * HW_ + hw];
        float dx = om[(b * 27 + 9 + p) * HW_ + hw];
        float mk = om[(b * 27 + 18 + p) * HW_ + hw];
        float py = dy + (float)(p / 3 - 1 + h);
        float px = dx + (float)(p % 3 - 1 + w);
        float fy = floorf(py), fx = floorf(px);
        int   y0 = (int)fy,   x0 = (int)fx;
        float wy = py - fy,   wx = px - fx;
        float cw[4] = { (1.f - wy) * (1.f - wx), (1.f - wy) * wx,
                        wy * (1.f - wx),         wy * wx };
        int cy[4] = { y0, y0, y0 + 1, y0 + 1 };
        int cx[4] = { x0, x0 + 1, x0, x0 + 1 };
#pragma unroll
        for (int cc = 0; cc < 4; ++cc) {
            bool valid = (cy[cc] >= 0) && (cy[cc] < HH_) && (cx[cc] >= 0) && (cx[cc] < WW_);
            int yc = imin(imax(cy[cc], 0), HH_ - 1);
            int xc = imin(imax(cx[cc], 0), WW_ - 1);
            midx[e * 4 + cc] = yc * WW_ + xc;
            mwt [e * 4 + cc] = valid ? (cw[cc] * mk) : 0.f;
        }
    }
    __syncthreads();

    const unsigned  ldsb = (unsigned)(uintptr_t)(void*)Btlds;
    v8f acc0 = {}; v8f acc1 = {};
    for (int k0 = 0; k0 < KDC_; k0 += 32) {
        // B: async global->LDS, one contiguous pre-tiled 20 KB chunk
        unsigned long long gb =
            (unsigned long long)(uintptr_t)(B2t + (size_t)(k0 >> 5) * (CO_ * LDB_));
#pragma unroll
        for (int it = 0; it < 5; ++it) {                   // 256*LDB_*2/16 = 1280 chunks
            int e = tid + it * 256;
            async_g2l_b128(ldsb + (unsigned)(e * 16), gb + (unsigned long long)(e * 16));
        }
        // A: bilinear gather (fp32) -> bf16
        for (int e = tid; e < 16 * 32; e += 256) {
            int r = e >> 5, kk = e & 31;
            int k = k0 + kk;
            int c = k / 9, p = k - 9 * c;
            int base = (mb[r] * CI_ + c) * HW_;
            const int*   ip = &midx[(r * 9 + p) * 4];
            const float* wp = &mwt [(r * 9 + p) * 4];
            float v = x[base + ip[0]] * wp[0] + x[base + ip[1]] * wp[1]
                    + x[base + ip[2]] * wp[2] + x[base + ip[3]] * wp[3];
            Alds[r * LDA_ + kk] = f2bf(v);
        }
        wait_async0();
        __syncthreads();
        FragBf fa, fb0, fb1;
        load_frag_a(Alds, LDA_, fa);
        load_frag_b(Btlds + (wave * 32) * LDB_, LDB_, fb0);
        load_frag_b(Btlds + (wave * 32 + 16) * LDB_, LDB_, fb1);
        acc0 = wmma_bf16(fa, fb0, acc0);
        acc1 = wmma_bf16(fa, fb1, acc1);
        __syncthreads();
    }
    // BN1 + ReLU, write y (fp32) + y_bf16
#pragma unroll
    for (int half = 0; half < 2; ++half) {
        int o = wave * 32 + half * 16 + (lane & 15);
        float sc = g1[o] * rsqrtf(va1[o] + 1e-5f);
        float sh = be1[o] - mu1[o] * sc;
        v8f& acc = half ? acc1 : acc0;
#pragma unroll
        for (int i = 0; i < 8; ++i) {
            int m  = m0 + ((lane >> 4) * 8 + i);
            int b  = m >> 12, hw = m & 4095;
            float v = acc[i] * sc + sh;
            v = v > 0.f ? v : 0.f;
            int idx = (b * CO_ + o) * HW_ + hw;
            y_out[idx] = v;
            y_bf[idx]  = f2bf(v);
        }
    }
}

// ---------------- Kernel 3: 4x4 stride-2 transposed conv + BN2 + ReLU ----------------
// Per phase (pY,pX): GEMM M=32768, N=256, K=1024.  grid (2048, 4).
__global__ __launch_bounds__(256)
void k_deconv(const unsigned short* __restrict__ y_bf, const unsigned short* __restrict__ B3t,
              const float* __restrict__ g2, const float* __restrict__ be2,
              const float* __restrict__ mu2, const float* __restrict__ va2,
              float* __restrict__ up_out) {
    __shared__ __align__(16) unsigned short Alds[16 * LDA_];
    __shared__ __align__(16) unsigned short Btlds[256 * LDB_];
    __shared__ int t_idx[16 * 4];
    __shared__ int t_b[16];

    const int tid  = threadIdx.x;
    const int wave = tid >> 5;
    const int lane = tid & 31;
    const int m0   = blockIdx.x * 16;
    const int ph   = blockIdx.y;
    const int pY   = ph >> 1, pX = ph & 1;

    for (int e = tid; e < 16 * 4; e += 256) {
        int r = e >> 2, t = e & 3, ty = t >> 1, tx = t & 1;
        int m = m0 + r;
        int b = m >> 12, hw = m & 4095;
        int oy = hw >> 6, ox = hw & 63;
        if (t == 0) t_b[r] = b;
        int h = oy + ty + pY - 1;
        int w = ox + tx + pX - 1;
        t_idx[e] = (h >= 0 && h < HH_ && w >= 0 && w < WW_) ? (h * WW_ + w) : -1;
    }
    __syncthreads();

    const unsigned ldsb = (unsigned)(uintptr_t)(void*)Btlds;
    const unsigned short* Bph = B3t + (size_t)ph * (KUP_ / 32) * (CO_ * LDB_);
    v8f acc0 = {}; v8f acc1 = {};
    for (int k0 = 0; k0 < KUP_; k0 += 32) {
        unsigned long long gb =
            (unsigned long long)(uintptr_t)(Bph + (size_t)(k0 >> 5) * (CO_ * LDB_));
#pragma unroll
        for (int it = 0; it < 5; ++it) {
            int e = tid + it * 256;
            async_g2l_b128(ldsb + (unsigned)(e * 16), gb + (unsigned long long)(e * 16));
        }
        for (int e = tid; e < 16 * 32; e += 256) {
            int r = e >> 5, kk = e & 31;
            int k = k0 + kk;
            int c = k >> 2, t = k & 3;
            int sidx = t_idx[r * 4 + t];
            Alds[r * LDA_ + kk] =
                (sidx >= 0) ? y_bf[(t_b[r] * CO_ + c) * HW_ + sidx] : (unsigned short)0;
        }
        wait_async0();
        __syncthreads();
        FragBf fa, fb0, fb1;
        load_frag_a(Alds, LDA_, fa);
        load_frag_b(Btlds + (wave * 32) * LDB_, LDB_, fb0);
        load_frag_b(Btlds + (wave * 32 + 16) * LDB_, LDB_, fb1);
        acc0 = wmma_bf16(fa, fb0, acc0);
        acc1 = wmma_bf16(fa, fb1, acc1);
        __syncthreads();
    }
#pragma unroll
    for (int half = 0; half < 2; ++half) {
        int o = wave * 32 + half * 16 + (lane & 15);
        float sc = g2[o] * rsqrtf(va2[o] + 1e-5f);
        float sh = be2[o] - mu2[o] * sc;
        v8f& acc = half ? acc1 : acc0;
#pragma unroll
        for (int i = 0; i < 8; ++i) {
            int m  = m0 + ((lane >> 4) * 8 + i);
            int b  = m >> 12, hw = m & 4095;
            int oy = hw >> 6, ox = hw & 63;
            int Y = 2 * oy + pY, X = 2 * ox + pX;
            float v = acc[i] * sc + sh;
            v = v > 0.f ? v : 0.f;
            up_out[((size_t)b * CO_ + o) * (128 * 128) + (size_t)Y * 128 + X] = v;
        }
    }
}

// ---------------- Host launcher ----------------
extern "C" void kernel_launch(void* const* d_in, const int* in_sizes, int n_in,
                              void* d_out, int out_size, void* d_ws, size_t ws_size,
                              hipStream_t stream) {
    const float* x     = (const float*)d_in[0];
    const float* w_off = (const float*)d_in[1];
    const float* b_off = (const float*)d_in[2];
    const float* w_dcn = (const float*)d_in[3];
    const float* g1    = (const float*)d_in[4];
    const float* be1   = (const float*)d_in[5];
    const float* mu1   = (const float*)d_in[6];
    const float* va1   = (const float*)d_in[7];
    const float* w_up  = (const float*)d_in[8];
    const float* g2    = (const float*)d_in[9];
    const float* be2   = (const float*)d_in[10];
    const float* mu2   = (const float*)d_in[11];
    const float* va2   = (const float*)d_in[12];

    float* y_out  = (float*)d_out;                            // (8,256,64,64)
    float* up_out = (float*)d_out + (size_t)BN_ * CO_ * HW_;  // (8,256,128,128)

    char* ws = (char*)d_ws;
    float*          om  = (float*)ws;                                   //  3,538,944 B
    unsigned short* yb  = (unsigned short*)(ws + 3538944);              // 16,777,216 B
    unsigned short* B1  = (unsigned short*)(ws + 3538944 + 16777216);   //    147,456 B
    unsigned short* B2t = (unsigned short*)(ws + 3538944 + 16777216 + 147456);   // 1,474,560 B
    unsigned short* B3t = (unsigned short*)(ws + 3538944 + 16777216 + 147456 + 1474560); // 2,621,440 B

    k_pack_woff<<<(KDC_ * 32 + 255) / 256, 256, 0, stream>>>(w_off, B1);
    k_pack_wdcn<<<((KDC_ / 32) * CO_ * LDB_ + 255) / 256, 256, 0, stream>>>(w_dcn, B2t);
    k_pack_wup <<<(4 * (KUP_ / 32) * CO_ * LDB_ + 255) / 256, 256, 0, stream>>>(w_up, B3t);

    k_offset_conv<<<MTOT_ / 128, 256, 0, stream>>>(x, B1, b_off, om);

    k_deform<<<MTOT_ / 16, 256, 0, stream>>>(x, om, B2t, g1, be1, mu1, va1, y_out, yb);

    k_deconv<<<dim3(MTOT_ / 16, 4), 256, 0, stream>>>(yb, B3t, g2, be2, mu2, va2, up_out);
}
